// SteinKernel_77807627534749
// MI455X (gfx1250) — compile-verified, matrix operand order checked by reading, not run on hardware
//
#include <hip/hip_runtime.h>
#include <hip/hip_bf16.h>
#include <math.h>

typedef float  v8f  __attribute__((ext_vector_type(8)));
typedef __bf16 v4bf __attribute__((ext_vector_type(4)));
typedef __bf16 v8bf __attribute__((ext_vector_type(8)));
typedef __bf16 v16bf __attribute__((ext_vector_type(16)));

#define B_ROWS 4096
#define D_DIM  512
#define SAMPLE 512

// workspace float offsets
#define XN_OFF    0
#define YN_OFF    4096
#define YSY_OFF   8192
#define D2_OFF    12288            // 512*512 = 262144 floats
#define PARAM_OFF 274432           // 4 floats

// ---------------------------------------------------------------------------
// Kernel 1: per-row stats: xn[i]=||x_i||^2, yn[j]=||y_j||^2, ysy[j]=<y_j,sy_j>
// ---------------------------------------------------------------------------
__global__ void rowstats_kernel(const float* __restrict__ x,
                                const float* __restrict__ y,
                                const float* __restrict__ sy,
                                float* __restrict__ xn,
                                float* __restrict__ yn,
                                float* __restrict__ ysy) {
    int row  = blockIdx.x * (blockDim.x >> 5) + (threadIdx.x >> 5);
    int lane = threadIdx.x & 31;
    if (row >= B_ROWS) return;
    const size_t base = (size_t)row * D_DIM;
    float sxx = 0.f, syy = 0.f, sdot = 0.f;
    for (int c = lane; c < D_DIM; c += 32) {
        float vx = x[base + c];
        float vy = y[base + c];
        float vs = sy[base + c];
        sxx  += vx * vx;
        syy  += vy * vy;
        sdot += vy * vs;
    }
#pragma unroll
    for (int off = 16; off > 0; off >>= 1) {
        sxx  += __shfl_xor(sxx,  off, 32);
        syy  += __shfl_xor(syy,  off, 32);
        sdot += __shfl_xor(sdot, off, 32);
    }
    if (lane == 0) { xn[row] = sxx; yn[row] = syy; ysy[row] = sdot; }
}

// ---------------------------------------------------------------------------
// Kernel 2: sample pairwise dist_sq for the first 512 rows (median heuristic)
// ---------------------------------------------------------------------------
__global__ void sample_dist_kernel(const float* __restrict__ x,
                                   const float* __restrict__ y,
                                   const float* __restrict__ xn,
                                   const float* __restrict__ yn,
                                   float* __restrict__ d2) {
    __shared__ float Ax[16][17];
    __shared__ float Ay[16][17];
    const int tx = threadIdx.x, ty = threadIdx.y;
    const int i = blockIdx.y * 16 + ty;
    const int j = blockIdx.x * 16 + tx;
    float acc = 0.f;
    for (int c0 = 0; c0 < D_DIM; c0 += 16) {
        Ax[ty][tx] = x[(size_t)i * D_DIM + c0 + tx];
        Ay[ty][tx] = y[(size_t)(blockIdx.x * 16 + ty) * D_DIM + c0 + tx];
        __syncthreads();
#pragma unroll
        for (int c = 0; c < 16; ++c) acc += Ax[ty][c] * Ay[tx][c];
        __syncthreads();
    }
    float d = xn[i] + yn[j] - 2.f * acc;
    d2[(size_t)i * SAMPLE + j] = fmaxf(d, 0.f);
}

// ---------------------------------------------------------------------------
// Kernel 3: exact median via 4-pass MSB radix select on float bit patterns
// (non-negative floats order-preserve as uint32), then bandwidth constants.
// ---------------------------------------------------------------------------
__global__ void median_params_kernel(const float* __restrict__ d2,
                                     float* __restrict__ params) {
    __shared__ unsigned hist[256];
    __shared__ unsigned sel_byte;
    __shared__ unsigned sel_k;
    const unsigned t = threadIdx.x;      // blockDim = 1024
    const unsigned N = SAMPLE * SAMPLE;  // 262144
    unsigned vsel[2];

    for (int which = 0; which < 2; ++which) {
        unsigned target = 131071u + (unsigned)which;
        unsigned prefix = 0, mask = 0;
        for (int p = 3; p >= 0; --p) {
            if (t < 256) hist[t] = 0u;
            __syncthreads();
            for (unsigned idx = t; idx < N; idx += 1024u) {
                unsigned bits = __float_as_uint(d2[idx]);
                if ((bits & mask) == prefix)
                    atomicAdd(&hist[(bits >> (8 * p)) & 255u], 1u);
            }
            __syncthreads();
            if (t == 0) {
                unsigned cum = 0, b = 0;
                for (b = 0; b < 256; ++b) {
                    unsigned h = hist[b];
                    if (cum + h > target) break;
                    cum += h;
                }
                sel_byte = b;
                sel_k = target - cum;
            }
            __syncthreads();
            prefix |= sel_byte << (8 * p);
            mask   |= 255u << (8 * p);
            target  = sel_k;
            __syncthreads();
        }
        vsel[which] = prefix;
    }
    if (t == 0) {
        float med = 0.5f * (__uint_as_float(vsel[0]) + __uint_as_float(vsel[1]));
        float h = sqrtf(med * 0.5f);
        h = fminf(fmaxf(h, 0.1f), 10.0f);
        float h2 = h * h;
        params[0] = 1.0f / (2.0f * h2);          // inv2h2
        params[1] = 1.0f / h2;                   // invh2
        params[2] = 1.0f / (h2 * h2);            // invh4
        params[3] = (float)D_DIM / h2;           // D/h^2
    }
}

// ---------------------------------------------------------------------------
// Kernel 4: fused triple-GEMM + Stein epilogue, 3xBF16-split on
// V_WMMA_F32_16X16X32_BF16 (x = hi + lo bf16 planes; x.y ~ hh + hl + lh,
// fp32 accumulate). Workgroup tile 128(M)x64(N), 8 waves, each a 32x32
// micro-tile (2x2 WMMA positions) with 3 accumulator sets.
// ---------------------------------------------------------------------------
#define BM    128
#define BN    64
#define KB    64        // K staged per LDS block (2 WMMA K-steps)
#define LDSTB 72        // bf16 row stride (144B): 16B-aligned, conflict-free

__device__ __forceinline__ v16bf ldfrag(const __bf16* base, int rowoff, int kc) {
    v8bf c0 = *(const v8bf*)(base + rowoff + kc);        // K chunk 8*half .. +7
    v8bf c1 = *(const v8bf*)(base + rowoff + kc + 16);   // K chunk 16+8*half .. +7
    return __builtin_shufflevector(c0, c1, 0, 1, 2, 3, 4, 5, 6, 7,
                                           8, 9, 10, 11, 12, 13, 14, 15);
}

__device__ __forceinline__ v8f wmma_bf(v16bf a, v16bf b, v8f c) {
    return __builtin_amdgcn_wmma_f32_16x16x32_bf16(false, a, false, b,
                                                   (short)0, c, false, false);
}

// split one float4 into hi/lo bf16 quads and store to LDS planes
__device__ __forceinline__ void split_store(const float* __restrict__ src,
                                            __bf16* __restrict__ hi,
                                            __bf16* __restrict__ lo,
                                            int idx) {
    float4 v = *(const float4*)src;
    __bf16 h0 = (__bf16)v.x, h1 = (__bf16)v.y, h2 = (__bf16)v.z, h3 = (__bf16)v.w;
    v4bf hv = {h0, h1, h2, h3};
    v4bf lv = {(__bf16)(v.x - (float)h0), (__bf16)(v.y - (float)h1),
               (__bf16)(v.z - (float)h2), (__bf16)(v.w - (float)h3)};
    *(v4bf*)&hi[idx] = hv;
    *(v4bf*)&lo[idx] = lv;
}

__global__ __launch_bounds__(256, 1)
void stein_main_kernel(const float* __restrict__ x,
                       const float* __restrict__ sx,
                       const float* __restrict__ y,
                       const float* __restrict__ sy,
                       const float* __restrict__ stats,   // ws base: xn|yn|ysy
                       const float* __restrict__ params,  // inv2h2, invh2, invh4, D/h2
                       float* __restrict__ out) {
    __shared__ __align__(16) __bf16 sXhi [BM * LDSTB];
    __shared__ __align__(16) __bf16 sXlo [BM * LDSTB];
    __shared__ __align__(16) __bf16 sSXhi[BM * LDSTB];
    __shared__ __align__(16) __bf16 sSXlo[BM * LDSTB];
    __shared__ __align__(16) __bf16 sYhi [BN * LDSTB];
    __shared__ __align__(16) __bf16 sYlo [BN * LDSTB];
    __shared__ __align__(16) __bf16 sSYhi[BN * LDSTB];
    __shared__ __align__(16) __bf16 sSYlo[BN * LDSTB];

    const int tid  = threadIdx.x;
    const int lane = tid & 31;
    const int wave = tid >> 5;
    const int wr   = wave >> 1;   // 0..3 : 32 M-rows each
    const int wc   = wave & 1;    // 0..1 : 32 N-cols each
    const int l16  = lane & 15;
    const int half = lane >> 4;

    const int m0 = blockIdx.y * BM;
    const int n0 = blockIdx.x * BN;

    const v8f vzero = {0.f, 0.f, 0.f, 0.f, 0.f, 0.f, 0.f, 0.f};
    v8f aS1[2][2], aS2[2][2], aS3[2][2];
#pragma unroll
    for (int mb = 0; mb < 2; ++mb)
#pragma unroll
        for (int nb = 0; nb < 2; ++nb) {
            aS1[mb][nb] = vzero; aS2[mb][nb] = vzero; aS3[mb][nb] = vzero;
        }

    for (int kb = 0; kb < D_DIM; kb += KB) {
        // ---- stage K-block: fp32 -> hi/lo bf16 planes in LDS
#pragma unroll
        for (int it = 0; it < 8; ++it) {   // X, SX: 128 rows x 64 cols = 2048 f4
            int c   = tid + it * 256;
            int row = c >> 4;              // 16 float4 per row
            int c4  = (c & 15) << 2;
            int di  = row * LDSTB + c4;
            split_store(&x [(size_t)(m0 + row) * D_DIM + kb + c4], sXhi,  sXlo,  di);
            split_store(&sx[(size_t)(m0 + row) * D_DIM + kb + c4], sSXhi, sSXlo, di);
        }
#pragma unroll
        for (int it = 0; it < 4; ++it) {   // Y, SY: 64 rows x 64 cols = 1024 f4
            int c   = tid + it * 256;
            int row = c >> 4;
            int c4  = (c & 15) << 2;
            int di  = row * LDSTB + c4;
            split_store(&y [(size_t)(n0 + row) * D_DIM + kb + c4], sYhi,  sYlo,  di);
            split_store(&sy[(size_t)(n0 + row) * D_DIM + kb + c4], sSYhi, sSYlo, di);
        }
        __syncthreads();

        // ---- compute: 2 K=32 steps, 36 WMMAs each (3 products x 4 tiles x 3 split terms)
#pragma unroll
        for (int k32 = 0; k32 < KB; k32 += 32) {
            const int kc  = k32 + (half << 3);
            const int ra0 = (wr * 32      + l16) * LDSTB;
            const int ra1 = (wr * 32 + 16 + l16) * LDSTB;
            const int rb0 = (wc * 32      + l16) * LDSTB;
            const int rb1 = (wc * 32 + 16 + l16) * LDSTB;

            v16bf axh0 = ldfrag(sXhi, ra0, kc), axl0 = ldfrag(sXlo, ra0, kc);
            v16bf axh1 = ldfrag(sXhi, ra1, kc), axl1 = ldfrag(sXlo, ra1, kc);

            // ---- S1 = x . y^T
            {
                v16bf byh0 = ldfrag(sYhi, rb0, kc), byl0 = ldfrag(sYlo, rb0, kc);
                v16bf byh1 = ldfrag(sYhi, rb1, kc), byl1 = ldfrag(sYlo, rb1, kc);
                aS1[0][0] = wmma_bf(axh0, byh0, wmma_bf(axl0, byh0, wmma_bf(axh0, byl0, aS1[0][0])));
                aS1[0][1] = wmma_bf(axh0, byh1, wmma_bf(axl0, byh1, wmma_bf(axh0, byl1, aS1[0][1])));
                aS1[1][0] = wmma_bf(axh1, byh0, wmma_bf(axl1, byh0, wmma_bf(axh1, byl0, aS1[1][0])));
                aS1[1][1] = wmma_bf(axh1, byh1, wmma_bf(axl1, byh1, wmma_bf(axh1, byl1, aS1[1][1])));
            }

            // ---- S3 = x . sy^T (reuses a_x fragments)
            v16bf bsh0 = ldfrag(sSYhi, rb0, kc), bsl0 = ldfrag(sSYlo, rb0, kc);
            v16bf bsh1 = ldfrag(sSYhi, rb1, kc), bsl1 = ldfrag(sSYlo, rb1, kc);
            aS3[0][0] = wmma_bf(axh0, bsh0, wmma_bf(axl0, bsh0, wmma_bf(axh0, bsl0, aS3[0][0])));
            aS3[0][1] = wmma_bf(axh0, bsh1, wmma_bf(axl0, bsh1, wmma_bf(axh0, bsl1, aS3[0][1])));
            aS3[1][0] = wmma_bf(axh1, bsh0, wmma_bf(axl1, bsh0, wmma_bf(axh1, bsl0, aS3[1][0])));
            aS3[1][1] = wmma_bf(axh1, bsh1, wmma_bf(axl1, bsh1, wmma_bf(axh1, bsl1, aS3[1][1])));

            // ---- S2 = sx . sy^T (reuses b_sy fragments; a_x now dead)
            {
                v16bf ash0 = ldfrag(sSXhi, ra0, kc), asl0 = ldfrag(sSXlo, ra0, kc);
                v16bf ash1 = ldfrag(sSXhi, ra1, kc), asl1 = ldfrag(sSXlo, ra1, kc);
                aS2[0][0] = wmma_bf(ash0, bsh0, wmma_bf(asl0, bsh0, wmma_bf(ash0, bsl0, aS2[0][0])));
                aS2[0][1] = wmma_bf(ash0, bsh1, wmma_bf(asl0, bsh1, wmma_bf(ash0, bsl1, aS2[0][1])));
                aS2[1][0] = wmma_bf(ash1, bsh0, wmma_bf(asl1, bsh0, wmma_bf(ash1, bsl0, aS2[1][0])));
                aS2[1][1] = wmma_bf(ash1, bsh1, wmma_bf(asl1, bsh1, wmma_bf(ash1, bsl1, aS2[1][1])));
            }
        }
        __syncthreads();
    }

    // ---- epilogue: dist, exp, Stein terms; each output written once
    const float p_inv2h2 = params[0];
    const float p_invh2  = params[1];
    const float p_invh4  = params[2];
    const float p_kD     = params[3];

#pragma unroll
    for (int mb = 0; mb < 2; ++mb) {
#pragma unroll
        for (int nb = 0; nb < 2; ++nb) {
            const int j = n0 + wc * 32 + nb * 16 + l16;
            const float ynrm = stats[YN_OFF + j];
            const float ysyj = stats[YSY_OFF + j];
#pragma unroll
            for (int r = 0; r < 8; ++r) {
                const int i = m0 + wr * 32 + mb * 16 + half * 8 + r;
                const float xnrm = stats[XN_OFF + i];
                const float s1 = aS1[mb][nb][r];
                const float s2 = aS2[mb][nb][r];
                const float s3 = aS3[mb][nb][r];
                float dist = fmaxf(xnrm + ynrm - 2.0f * s1, 0.0f);
                float kv   = __expf(-dist * p_inv2h2);
                float res  = kv * s2
                           + (s3 - ysyj) * kv * p_invh2
                           + dist * kv * p_invh4
                           - kv * p_kD;
                out[(size_t)i * B_ROWS + j] = res;
            }
        }
    }
}

// ---------------------------------------------------------------------------
extern "C" void kernel_launch(void* const* d_in, const int* in_sizes, int n_in,
                              void* d_out, int out_size, void* d_ws, size_t ws_size,
                              hipStream_t stream) {
    const float* x  = (const float*)d_in[0];
    const float* sx = (const float*)d_in[1];
    const float* y  = (const float*)d_in[2];
    const float* sy = (const float*)d_in[3];
    float* out = (float*)d_out;
    float* wsf = (float*)d_ws;

    rowstats_kernel<<<512, 256, 0, stream>>>(x, y, sy,
                                             wsf + XN_OFF, wsf + YN_OFF, wsf + YSY_OFF);

    dim3 g2(SAMPLE / 16, SAMPLE / 16), b2(16, 16);
    sample_dist_kernel<<<g2, b2, 0, stream>>>(x, y, wsf + XN_OFF, wsf + YN_OFF,
                                              wsf + D2_OFF);

    median_params_kernel<<<1, 1024, 0, stream>>>(wsf + D2_OFF, wsf + PARAM_OFF);

    dim3 g4(B_ROWS / BN, B_ROWS / BM);   // (64, 32)
    stein_main_kernel<<<g4, 256, 0, stream>>>(x, sx, y, sy, wsf, wsf + PARAM_OFF, out);
}
